// MultiHeadAttention_91207925498448
// MI455X (gfx1250) — matrix operand order repaired and checked by reading, not compiled, not run on hardware
//
#include <hip/hip_runtime.h>
#include <hip/hip_bf16.h>

// MHA: B=2, S=2048, D=1024, H=16, d_k=64.
// bf16 WMMA (v_wmma_f32_16x16x32_bf16) + flash attention with async
// global->LDS staging (ASYNCcnt). LDS tile layouts chosen so every WMMA
// fragment is a contiguous per-lane vector load (ds_load_b128).

typedef __attribute__((ext_vector_type(16))) __bf16 v16bf;
typedef __attribute__((ext_vector_type(8)))  __bf16 v8bf;
typedef __attribute__((ext_vector_type(8)))  float  v8f;
typedef int v4i __attribute__((vector_size(16)));   // int4, matches builtin sig

#define DMODEL 1024
#define NH     16
#define DK     64
#define SEQ    2048
#define NB     2
#define MROWS  (NB * SEQ)   // 4096

// global->LDS async copy, 16 bytes per lane (tracked by ASYNCcnt).
#define ASYNC_CP_B128(src_glob, dst_lds)                                      \
  __builtin_amdgcn_global_load_async_to_lds_b128(                             \
      (__attribute__((address_space(1))) v4i*)(src_glob),                     \
      (__attribute__((address_space(3))) v4i*)(dst_lds), 0, 0)

static __device__ __forceinline__ v8f wmma_bf16(v16bf a, v16bf b, v8f c) {
  return __builtin_amdgcn_wmma_f32_16x16x32_bf16(
      /*neg_a=*/false, a, /*neg_b=*/false, b,
      /*c_mod=*/(short)0, c, /*reuse_a=*/false, /*reuse_b=*/false);
}

// A-fragment k index (16-bit A, 16x32, wave32):
//   lanes 0-15: V0..3 -> K=0..7, V4..7 -> K=16..23
//   lanes16-31: V0..3 -> K=8..15, V4..7 -> K=24..31
static __device__ __forceinline__ int kmapA(int e, int hi) {
  return e + ((e >= 8) ? 8 : 0) + (hi ? 8 : 0);
}

static __device__ __forceinline__ v16bf cat8(v8bf lo, v8bf hi8) {
  return __builtin_shufflevector(lo, hi8, 0, 1, 2, 3, 4, 5, 6, 7,
                                 8, 9, 10, 11, 12, 13, 14, 15);
}
// A fragment: two contiguous 8-element runs at k = 8*hi and k = 16 + 8*hi.
static __device__ __forceinline__ v16bf loadA16(const __bf16* row, int hi) {
  const v8bf lo  = *(const v8bf*)(row + 8 * hi);
  const v8bf hh  = *(const v8bf*)(row + 16 + 8 * hi);
  return cat8(lo, hh);
}
// B fragment: two contiguous 8-element runs at k = 8*hi and k = 16 + 8*hi.
static __device__ __forceinline__ v16bf loadB16(const __bf16* row, int hi) {
  const v8bf lo  = *(const v8bf*)(row + 8 * hi);
  const v8bf hh  = *(const v8bf*)(row + 16 + 8 * hi);
  return cat8(lo, hh);
}

// ---------------------------------------------------------------------------
// Kernel 0: convert the four weight matrices to bf16 once (each is re-read
// ~256x by the GEMMs; bf16 halves that stream and removes per-use cvts).
// ---------------------------------------------------------------------------
__global__ void __launch_bounds__(256)
cvt_w_kernel(const float* __restrict__ w0, const float* __restrict__ w1,
             const float* __restrict__ w2, const float* __restrict__ w3,
             __bf16* __restrict__ dst) {
  const int idx = blockIdx.x * 256 + threadIdx.x;      // 0 .. 1M-1
  const int z   = blockIdx.y;                          // which matrix
  const float* s = (z == 0) ? w0 : (z == 1) ? w1 : (z == 2) ? w2 : w3;
  dst[(size_t)z * DMODEL * DMODEL + idx] = (__bf16)s[idx];
}

// ---------------------------------------------------------------------------
// Kernel 1: fused QKV projection.  out = X @ W^T + b  (Linear semantics).
// One wave computes a 16x64 strip (4 accumulators reusing one A fragment).
// z = 0 -> Q stored [B,H,S,DK]  bf16 (natural)
// z = 1 -> K stored [B,H,S,DK]  bf16 (natural: score-tile is staged via LDS)
// z = 2 -> V stored [B,H,DK,S]  bf16 (transposed: PV-GEMM B operand)
// ---------------------------------------------------------------------------
__global__ void __launch_bounds__(256)
proj_kernel(const float* __restrict__ qin, const float* __restrict__ kin,
            const float* __restrict__ vin, const __bf16* __restrict__ Wall,
            const float* __restrict__ bq, const float* __restrict__ bk,
            const float* __restrict__ bv,
            __bf16* __restrict__ Qb, __bf16* __restrict__ Kb,
            __bf16* __restrict__ Vt) {
  const int lane = threadIdx.x & 31;
  const int wave = threadIdx.x >> 5;
  const int hi   = lane >> 4;
  const int l16  = lane & 15;
  const int z    = blockIdx.z;

  const float*  X  = (z == 0) ? qin : (z == 1) ? kin : vin;
  const float*  B  = (z == 0) ? bq  : (z == 1) ? bk  : bv;
  const __bf16* Wb = Wall + (size_t)z * DMODEL * DMODEL;

  const int n0 = (blockIdx.x * 8 + wave) * 64;  // 64-wide n strip
  const int m0 = blockIdx.y * 16;               // token tile (flat B*S)

  v8f acc[4] = {};
  for (int k0 = 0; k0 < DMODEL; k0 += 32) {
    __builtin_prefetch(&X[(m0 + l16) * DMODEL + k0 + 128], 0, 1);
    v16bf a;
#pragma unroll
    for (int e = 0; e < 16; ++e)
      a[e] = (__bf16)X[(m0 + l16) * DMODEL + k0 + kmapA(e, hi)];
#pragma unroll
    for (int t = 0; t < 4; ++t) {
      v16bf bm = loadB16(&Wb[(size_t)(n0 + t * 16 + l16) * DMODEL + k0], hi);
      acc[t] = wmma_bf16(a, bm, acc[t]);
    }
  }

#pragma unroll
  for (int t = 0; t < 4; ++t) {
    const int n  = n0 + t * 16 + l16;
    const float bn = B[n];
    const int hh = n >> 6;          // head
    const int dd = n & 63;          // within-head feature
#pragma unroll
    for (int r = 0; r < 8; ++r) {
      const int mg = m0 + r + (hi ? 8 : 0);
      const int bb = mg >> 11;      // batch
      const int ss = mg & 2047;     // sequence position
      const float val = acc[t][r] + bn;
      if (z == 0)
        Qb[((size_t)(bb * NH + hh) * SEQ + ss) * DK + dd] = (__bf16)val;
      else if (z == 1)
        Kb[((size_t)(bb * NH + hh) * SEQ + ss) * DK + dd] = (__bf16)val;
      else
        Vt[((size_t)(bb * NH + hh) * DK + dd) * SEQ + ss] = (__bf16)val;
    }
  }
}

// ---------------------------------------------------------------------------
// Kernel 2: flash attention. 8 waves/block share one (b,h); per kv-step the
// block async-copies the K (32x64, kv-major) and V (64x32, d-major) bf16
// tiles into LDS, then every wave reads WMMA B fragments as two contiguous
// 16B runs per lane. Online softmax keeps scores out of HBM entirely.
// ---------------------------------------------------------------------------
__global__ void __launch_bounds__(256)
attn_kernel(const __bf16* __restrict__ Qb, const __bf16* __restrict__ Kb,
            const __bf16* __restrict__ Vt, const int* __restrict__ mask,
            __bf16* __restrict__ attn_out) {
  __shared__ __attribute__((aligned(32))) __bf16 Ktile[32 * 64]; // [kv][d] 4KB
  __shared__ __attribute__((aligned(32))) __bf16 Vtile[64 * 32]; // [d][kv] 4KB
  __shared__ __attribute__((aligned(32))) __bf16 pbuf[8][16][32];// P stage 8KB

  const int tid  = threadIdx.x;
  const int lane = tid & 31;
  const int wave = tid >> 5;
  const int hi   = lane >> 4;
  const int l16  = lane & 15;

  const int bh = blockIdx.y;
  const int b  = bh >> 4;
  const int h  = bh & 15;
  const int q0 = (blockIdx.x * 8 + wave) * 16;

  const __bf16* Qp = Qb + (size_t)bh * SEQ * DK;
  const __bf16* Kp = Kb + (size_t)bh * SEQ * DK;
  const __bf16* Vp = Vt + (size_t)bh * DK * SEQ;

  // Per-thread async-copy coordinates (16B chunks).
  const int krow = tid >> 3, kch = tid & 7;   // K: 32 kv rows x 8 chunks
  const int vrow = tid >> 2, vch = tid & 3;   // V: 64 d  rows x 4 chunks

  // Q A-fragments (d = 0..31 and 32..63), loaded once as vectors.
  const __bf16* qrow = Qp + (q0 + l16) * DK;
  v16bf aq0 = loadA16(qrow, hi);
  v16bf aq1 = loadA16(qrow + 32, hi);

  v8f o0 = {}, o1 = {}, o2 = {}, o3 = {};
  float mrow[8], lrow[8];
#pragma unroll
  for (int r = 0; r < 8; ++r) { mrow[r] = -3.0e38f; lrow[r] = 0.0f; }

  const float scale = 0.125f;  // 1/sqrt(64)

  for (int jb = 0; jb < SEQ; jb += 32) {
    __syncthreads();  // previous tiles fully consumed before overwrite

    // ---- cooperative async stage of K (kv-major) and V (d-major) tiles ----
    ASYNC_CP_B128(Kp + (size_t)(jb + krow) * DK + kch * 8,
                  &Ktile[krow * 64 + kch * 8]);
    ASYNC_CP_B128(Vp + (size_t)vrow * SEQ + jb + vch * 8,
                  &Vtile[vrow * 32 + vch * 8]);
    asm volatile("s_wait_asynccnt 0" ::: "memory");
    __syncthreads();

    // ---- scores: two 16x16 tiles (columns jb..jb+15, jb+16..jb+31) ----
    v8f s0 = {}, s1 = {};
#pragma unroll
    for (int nblk = 0; nblk < 2; ++nblk) {
      const __bf16* krp = &Ktile[(nblk * 16 + l16) * 64];
      v16bf bk0 = loadB16(krp, hi);        // d = 0..31
      v16bf bk1 = loadB16(krp + 32, hi);   // d = 32..63
      v8f s = {};
      s = wmma_bf16(aq0, bk0, s);
      s = wmma_bf16(aq1, bk1, s);
      if (nblk == 0) s0 = s; else s1 = s;
    }

    // ---- scale + mask ----
#pragma unroll
    for (int r = 0; r < 8; ++r) {
      const int row = q0 + r + (hi ? 8 : 0);
      const size_t mb = ((size_t)b * SEQ + row) * SEQ + jb;
      float v0 = s0[r] * scale;
      float v1 = s1[r] * scale;
      if (mask[mb + l16] == 0)      v0 = -1.0e9f;
      if (mask[mb + 16 + l16] == 0) v1 = -1.0e9f;
      s0[r] = v0;
      s1[r] = v1;
    }

    // ---- online softmax update (per C-fragment row group of 16 lanes) ----
#pragma unroll
    for (int r = 0; r < 8; ++r) {
      float vmax = fmaxf(s0[r], s1[r]);
#pragma unroll
      for (int off = 1; off < 16; off <<= 1)
        vmax = fmaxf(vmax, __shfl_xor(vmax, off, 32));
      const float mnew  = fmaxf(mrow[r], vmax);
      const float alpha = __expf(mrow[r] - mnew);
      mrow[r] = mnew;
      const float p0 = __expf(s0[r] - mnew);
      const float p1 = __expf(s1[r] - mnew);
      float ps = p0 + p1;
#pragma unroll
      for (int off = 1; off < 16; off <<= 1)
        ps += __shfl_xor(ps, off, 32);
      lrow[r] = lrow[r] * alpha + ps;
      o0[r] *= alpha; o1[r] *= alpha; o2[r] *= alpha; o3[r] *= alpha;
      const int rl = r + (hi ? 8 : 0);
      pbuf[wave][rl][l16]      = (__bf16)p0;
      pbuf[wave][rl][16 + l16] = (__bf16)p1;
    }

    // Intra-wave DS store->load ordering (wave-private LDS region).
    asm volatile("s_wait_dscnt 0" ::: "memory");

    // ---- re-read P as A-fragment (16x32 over kv) ----
    v16bf ap = loadA16(&pbuf[wave][l16][0], hi);

    // ---- O += P @ V : four 16-wide d chunks, V from LDS (d-major) ----
#pragma unroll
    for (int dc = 0; dc < 4; ++dc) {
      v16bf bvf = loadB16(&Vtile[(dc * 16 + l16) * 32], hi);
      if      (dc == 0) o0 = wmma_bf16(ap, bvf, o0);
      else if (dc == 1) o1 = wmma_bf16(ap, bvf, o1);
      else if (dc == 2) o2 = wmma_bf16(ap, bvf, o2);
      else              o3 = wmma_bf16(ap, bvf, o3);
    }
  }

  // ---- epilogue: normalize, write [B,S,D] bf16 (head-transposed back) ----
#pragma unroll
  for (int r = 0; r < 8; ++r) {
    const float inv = 1.0f / lrow[r];
    const int row = q0 + r + (hi ? 8 : 0);
    const size_t base = ((size_t)b * SEQ + row) * DMODEL + h * DK;
    attn_out[base + 0  + l16] = (__bf16)(o0[r] * inv);
    attn_out[base + 16 + l16] = (__bf16)(o1[r] * inv);
    attn_out[base + 32 + l16] = (__bf16)(o2[r] * inv);
    attn_out[base + 48 + l16] = (__bf16)(o3[r] * inv);
  }
}

// ---------------------------------------------------------------------------
// Kernel 3: output projection. out = attn @ Wo^T + bo -> fp32 d_out.
// ---------------------------------------------------------------------------
__global__ void __launch_bounds__(256)
out_proj_kernel(const __bf16* __restrict__ attn, const __bf16* __restrict__ Wob,
                const float* __restrict__ bo, float* __restrict__ out) {
  const int lane = threadIdx.x & 31;
  const int wave = threadIdx.x >> 5;
  const int hi   = lane >> 4;
  const int l16  = lane & 15;

  const int n0 = (blockIdx.x * 8 + wave) * 64;
  const int m0 = blockIdx.y * 16;

  v8f acc[4] = {};
  for (int k0 = 0; k0 < DMODEL; k0 += 32) {
    __builtin_prefetch(&attn[(size_t)(m0 + l16) * DMODEL + k0 + 128], 0, 1);
    v16bf a = loadA16(&attn[(size_t)(m0 + l16) * DMODEL + k0], hi);
#pragma unroll
    for (int t = 0; t < 4; ++t) {
      v16bf bm = loadB16(&Wob[(size_t)(n0 + t * 16 + l16) * DMODEL + k0], hi);
      acc[t] = wmma_bf16(a, bm, acc[t]);
    }
  }

#pragma unroll
  for (int t = 0; t < 4; ++t) {
    const float bn = bo[n0 + t * 16 + l16];
#pragma unroll
    for (int r = 0; r < 8; ++r)
      out[(size_t)(m0 + r + (hi ? 8 : 0)) * DMODEL + n0 + t * 16 + l16] =
          acc[t][r] + bn;
  }
}

// ---------------------------------------------------------------------------
extern "C" void kernel_launch(void* const* d_in, const int* in_sizes, int n_in,
                              void* d_out, int out_size, void* d_ws,
                              size_t ws_size, hipStream_t stream) {
  const float* q    = (const float*)d_in[0];
  const float* k    = (const float*)d_in[1];
  const float* v    = (const float*)d_in[2];
  const int*   mask = (const int*)  d_in[3];
  const float* wq   = (const float*)d_in[4];
  const float* bq   = (const float*)d_in[5];
  const float* wk   = (const float*)d_in[6];
  const float* bk   = (const float*)d_in[7];
  const float* wv   = (const float*)d_in[8];
  const float* bv   = (const float*)d_in[9];
  const float* wo   = (const float*)d_in[10];
  const float* bo   = (const float*)d_in[11];
  float* out = (float*)d_out;

  // Workspace (bf16): Q | K | Vt | attn (4M elems each) | Wall (4 x 1M elems)
  const size_t qkv_elems = (size_t)NB * NH * SEQ * DK;   // 4M
  __bf16* Qb   = (__bf16*)d_ws;
  __bf16* Kb   = Qb + qkv_elems;
  __bf16* Vt   = Kb + qkv_elems;
  __bf16* attn = Vt + qkv_elems;
  __bf16* Wall = attn + (size_t)MROWS * DMODEL;          // 40 MB total

  // 0) weights -> bf16 (wq, wk, wv, wo)
  {
    dim3 grid(DMODEL * DMODEL / 256, 4);
    hipLaunchKernelGGL(cvt_w_kernel, grid, dim3(256), 0, stream,
                       wq, wk, wv, wo, Wall);
  }
  // 1) QKV projections: grid = (n-strips/8, m-tiles, {Q,K,V})
  {
    dim3 grid(DMODEL / (64 * 8), MROWS / 16, 3);
    hipLaunchKernelGGL(proj_kernel, grid, dim3(256), 0, stream,
                       q, k, v, Wall, bq, bk, bv, Qb, Kb, Vt);
  }
  // 2) fused flash attention: grid = (q-tiles/8, B*H)
  {
    dim3 grid(SEQ / (16 * 8), NB * NH);
    hipLaunchKernelGGL(attn_kernel, grid, dim3(256), 0, stream,
                       Qb, Kb, Vt, mask, attn);
  }
  // 3) output projection
  {
    dim3 grid(DMODEL / (64 * 8), MROWS / 16);
    hipLaunchKernelGGL(out_proj_kernel, grid, dim3(256), 0, stream,
                       attn, Wall + (size_t)3 * DMODEL * DMODEL, bo, out);
  }
}
